// SpikingShallowNetwork_70866960384642
// MI455X (gfx1250) — matrix operand order repaired and checked by reading, not compiled, not run on hardware
//
#include <hip/hip_runtime.h>

// Spiking shallow net: nsp = sum_t heaviside(v_t > 1), v_t = v_{t-1} + spikes_t @ W^T - s_{t-1}
// NT=128 (2 WMMA tiles per wave sharing one A fragment -> expansion cost amortized),
// async global->LDS staging of the W tile (global_load_async_to_lds_b128 + s_wait_asynccnt).

#define NSP   32
#define NIN   784
#define NINP  800      // padded K (25 * 32)
#define NOUT  1024
#define BATCH 4096
#define SSTR  808      // LDS row stride in f16 (404 dwords, stride%64=20 -> conflict-free)
#define BT    32       // batch rows per block
#define NT    128      // nout cols per block

typedef __attribute__((ext_vector_type(16))) _Float16 v16h;
typedef __attribute__((ext_vector_type(8)))  _Float16 v8h;
typedef __attribute__((ext_vector_type(8)))  float    v8f;
typedef __attribute__((ext_vector_type(4)))  int      v4i;

#if defined(__has_builtin)
#if __has_builtin(__builtin_amdgcn_global_load_async_to_lds_b128) && \
    __has_builtin(__builtin_amdgcn_s_wait_asynccnt)
#define HAVE_ASYNC_LDS 1
#endif
#endif
#ifndef HAVE_ASYNC_LDS
#define HAVE_ASYNC_LDS 0
#endif

__device__ __forceinline__ unsigned int hash32(unsigned int x) {
    x *= 0x9E3779B9u; x ^= x >> 16;
    x *= 0x85EBCA6Bu; x ^= x >> 13;
    x *= 0xC2B2AE35u; x ^= x >> 16;
    return x;
}

// ---- Kernel 1: pack 32 Bernoulli draws per (b,k) into one u32 bitmask ----
__global__ __launch_bounds__(256) void snn_pack_spikes(const float* __restrict__ x,
                                                       unsigned int* __restrict__ masks) {
    int idx = blockIdx.x * 256 + threadIdx.x;          // over BATCH*NINP
    int b = idx / NINP;
    int k = idx - b * NINP;
    unsigned int m = 0u;
    if (k < NIN) {
        float p = x[(size_t)b * NIN + k];
        unsigned int st = hash32((unsigned int)idx ^ 0x0123457u);
        #pragma unroll
        for (int t = 0; t < NSP; ++t) {
            st = st * 1664525u + 1013904223u;          // LCG per step
            float u = (float)(st >> 8) * (1.0f / 16777216.0f);
            m |= (u < p ? 1u : 0u) << t;
        }
    }
    masks[idx] = m;                                    // zero in K-pad region
}

// ---- Kernel 2: W f32 [NOUT][NIN] -> f16 [NOUT][NINP] (zero-padded K) ----
__global__ __launch_bounds__(256) void snn_convert_w(const float* __restrict__ W,
                                                     _Float16* __restrict__ Wh) {
    int idx = blockIdx.x * 256 + threadIdx.x;          // over NOUT*NINP
    int n = idx / NINP;
    int k = idx - n * NINP;
    Wh[idx] = (k < NIN) ? (_Float16)W[(size_t)n * NIN + k] : (_Float16)0.0f;
}

// ---- Kernel 3: fused 32-step LIF with WMMA GEMM ----
__global__ __launch_bounds__(256) void snn_main(const unsigned int* __restrict__ masks,
                                               const _Float16* __restrict__ Wh,
                                               float* __restrict__ out) {
    extern __shared__ char smem_raw[];
    _Float16* sW = (_Float16*)smem_raw;                        // [NT][SSTR] f16
    _Float16* sA = (_Float16*)(smem_raw + NT * SSTR * 2);      // [BT][SSTR] f16

    const int tid  = threadIdx.x;
    const int lane = tid & 31;
    const int wave = tid >> 5;        // 8 waves
    const int r    = lane & 15;
    const int hi   = lane >> 4;
    const int wrow = wave & 1;        // 2 row groups of 16
    const int wcol = wave >> 1;       // 4 col groups of 16 (x2 via +64 second tile)

    const int rowBase = blockIdx.x * BT;
    const int colBase = blockIdx.y * NT;

    __builtin_prefetch(masks + (size_t)rowBase * NINP, 0, 1);  // global_prefetch_b8

    // Stage W tile into LDS once; reused by all 32 time steps.
    for (int c = tid; c < NT * (NINP / 8); c += 256) {
        int col = c / (NINP / 8);
        int kk  = (c - col * (NINP / 8)) * 8;
        const _Float16* gp = Wh + (size_t)(colBase + col) * NINP + kk;
        _Float16* lp = sW + col * SSTR + kk;
#if HAVE_ASYNC_LDS
        __builtin_amdgcn_global_load_async_to_lds_b128(
            (__attribute__((address_space(1))) v4i*)gp,
            (__attribute__((address_space(3))) v4i*)lp, 0, 0);
#else
        *(uint4*)lp = *(const uint4*)gp;
#endif
    }
#if HAVE_ASYNC_LDS
    __builtin_amdgcn_s_wait_asynccnt(0);
#endif
    __syncthreads();

    v8f v0, v1, ns0, ns1;
    #pragma unroll
    for (int i = 0; i < 8; ++i) { v0[i] = 0.0f; v1[i] = 0.0f; ns0[i] = 0.0f; ns1[i] = 0.0f; }

    const _Float16* sArow  = sA + (wrow * 16 + r) * SSTR;
    const _Float16* sWcol0 = sW + (wcol * 16 + r) * SSTR;
    const _Float16* sWcol1 = sW + (64 + wcol * 16 + r) * SSTR;

    for (int t = 0; t < NSP; ++t) {
        // Expand bit t of the masks into f16 {0,1} spike tile in LDS.
        for (int c = tid; c < BT * (NINP / 8); c += 256) {
            int row = c / (NINP / 8);
            int kk  = (c - row * (NINP / 8)) * 8;
            const uint4* mp = (const uint4*)(masks + (size_t)(rowBase + row) * NINP + kk);
            uint4 m0 = mp[0], m1 = mp[1];
            unsigned int p0 = (((m0.x >> t) & 1u) * 0x3C00u) | ((((m0.y >> t) & 1u) * 0x3C00u) << 16);
            unsigned int p1 = (((m0.z >> t) & 1u) * 0x3C00u) | ((((m0.w >> t) & 1u) * 0x3C00u) << 16);
            unsigned int p2 = (((m1.x >> t) & 1u) * 0x3C00u) | ((((m1.y >> t) & 1u) * 0x3C00u) << 16);
            unsigned int p3 = (((m1.z >> t) & 1u) * 0x3C00u) | ((((m1.w >> t) & 1u) * 0x3C00u) << 16);
            uint4 pk; pk.x = p0; pk.y = p1; pk.z = p2; pk.w = p3;
            *(uint4*)(sA + row * SSTR + kk) = pk;
        }
        __syncthreads();

        // LIF soft reset fused into WMMA C operand: C0 = v - heaviside(v - 1)
        v8f acc0, acc1;
        #pragma unroll
        for (int i = 0; i < 8; ++i) {
            acc0[i] = v0[i] - (v0[i] > 1.0f ? 1.0f : 0.0f);
            acc1[i] = v1[i] - (v1[i] > 1.0f ? 1.0f : 0.0f);
        }

        // GEMM: 25 k-steps, 2 WMMAs each (shared A fragment)
        #pragma unroll 5
        for (int kb = 0; kb < NINP; kb += 32) {
            // A 16x32 f16: lanes 0-15 K=kb+0..15 ; lanes 16-31 K=kb+16..31
            v8h alo = *(const v8h*)(sArow + kb + hi * 16);
            v8h ahi = *(const v8h*)(sArow + kb + hi * 16 + 8);
            v16h a = __builtin_shufflevector(alo, ahi, 0,1,2,3,4,5,6,7,8,9,10,11,12,13,14,15);
            // B 32x16 f16: lanes 0-15 K=kb+0..15 ; lanes 16-31 K=kb+16..31 (col = lane&15)
            v8h b0lo = *(const v8h*)(sWcol0 + kb + hi * 16);
            v8h b0hi = *(const v8h*)(sWcol0 + kb + hi * 16 + 8);
            v8h b1lo = *(const v8h*)(sWcol1 + kb + hi * 16);
            v8h b1hi = *(const v8h*)(sWcol1 + kb + hi * 16 + 8);
            v16h b0 = __builtin_shufflevector(b0lo, b0hi, 0,1,2,3,4,5,6,7,8,9,10,11,12,13,14,15);
            v16h b1 = __builtin_shufflevector(b1lo, b1hi, 0,1,2,3,4,5,6,7,8,9,10,11,12,13,14,15);
            acc0 = __builtin_amdgcn_wmma_f32_16x16x32_f16(false, a, false, b0,
                                                          (short)0, acc0, false, false);
            acc1 = __builtin_amdgcn_wmma_f32_16x16x32_f16(false, a, false, b1,
                                                          (short)0, acc1, false, false);
        }
        __syncthreads();   // protect sA before next step's overwrite

        v0 = acc0;
        v1 = acc1;
        #pragma unroll
        for (int i = 0; i < 8; ++i) {
            ns0[i] += (v0[i] > 1.0f ? 1.0f : 0.0f);    // heaviside(v - T1)
            ns1[i] += (v1[i] > 1.0f ? 1.0f : 0.0f);
        }
    }

    // C/D layout: VGPR i -> M = i + hi*8, N = lane&15
    const int colg0 = colBase + wcol * 16 + r;
    const int colg1 = colg0 + 64;
    const int rowg  = rowBase + wrow * 16 + hi * 8;
    #pragma unroll
    for (int i = 0; i < 8; ++i) {
        out[(size_t)(rowg + i) * NOUT + colg0] = ns0[i];
        out[(size_t)(rowg + i) * NOUT + colg1] = ns1[i];
    }
}

extern "C" void kernel_launch(void* const* d_in, const int* in_sizes, int n_in,
                              void* d_out, int out_size, void* d_ws, size_t ws_size,
                              hipStream_t stream) {
    const float* x = (const float*)d_in[0];   // [BATCH, NIN]
    const float* W = (const float*)d_in[1];   // [NOUT, NIN]
    float* out = (float*)d_out;               // [BATCH, NOUT]

    _Float16* Wh = (_Float16*)d_ws;                                       // NOUT*NINP f16
    unsigned int* masks = (unsigned int*)((char*)d_ws + (size_t)NOUT * NINP * 2); // BATCH*NINP u32

    snn_pack_spikes<<<(BATCH * NINP) / 256, 256, 0, stream>>>(x, masks);
    snn_convert_w<<<(NOUT * NINP) / 256, 256, 0, stream>>>(W, Wh);

    dim3 grid(BATCH / BT, NOUT / NT);
    size_t lds_bytes = (size_t)(NT + BT) * SSTR * 2;   // 258,560 B (<320 KB/WGP)
    snn_main<<<grid, 256, lds_bytes, stream>>>(masks, Wh, out);
}
